// EncoderGCN_89713276878907
// MI455X (gfx1250) — matrix operand order repaired and checked by reading, not compiled, not run on hardware
//
#include <hip/hip_runtime.h>
#include <hip/hip_bf16.h>

typedef __attribute__((ext_vector_type(16))) _Float16 v16h;
typedef __attribute__((ext_vector_type(8)))  _Float16 v8h;
typedef __attribute__((ext_vector_type(8)))  float    v8f;

#define GCN_NTOT  32768
#define GCN_E     524288
#define GCN_G     64
#define GCN_N     512
#define GCN_D     256
#define GCN_FIN   128

// ---------------------------------------------------------------------------
// WMMA GEMM: C[M,256] = A[M,K](f16) x Bt[256,K](f16, pre-transposed) + bias
// Block = 256 threads (8 waves). Block handles a 16-row strip; wave w handles
// columns [w*32, w*32+32) as two 16x16 WMMA tiles sharing one A fragment.
// ---------------------------------------------------------------------------
__global__ void gemm_wmma_f16(const _Float16* __restrict__ A,
                              const _Float16* __restrict__ Bt,
                              const float* __restrict__ bias,
                              float* __restrict__ C,
                              int K) {
    const int lane   = threadIdx.x & 31;
    const int wave   = threadIdx.x >> 5;
    const int rowBase = blockIdx.x * 16;
    const int l15    = lane & 15;
    const int kgrpA  = (lane >> 4) * 16;   // A fragment K sub-base
    const int kgrpB  = (lane >> 4) * 16;   // B fragment K sub-base
    const int col0   = wave * 32;

    v8f acc0 = {};
    v8f acc1 = {};

    const _Float16* arowBase = A  + (size_t)(rowBase + l15) * K;
    const _Float16* brow0    = Bt + (size_t)(col0 + l15)      * K;
    const _Float16* brow1    = Bt + (size_t)(col0 + 16 + l15) * K;

    for (int k0 = 0; k0 < K; k0 += 32) {
        // A fragment: rows striped in lane, two contiguous 8-half chunks
        const _Float16* ap = arowBase + k0 + kgrpA;
        v8h alo = *(const v8h*)(ap);
        v8h ahi = *(const v8h*)(ap + 8);
        v16h a;
        #pragma unroll
        for (int i = 0; i < 8; ++i) { a[i] = alo[i]; a[i + 8] = ahi[i]; }

        // B fragments: column-contiguous (transposed weight), 16 contiguous halves
        const _Float16* bp0 = brow0 + k0 + kgrpB;
        const _Float16* bp1 = brow1 + k0 + kgrpB;
        v8h b0lo = *(const v8h*)(bp0);
        v8h b0hi = *(const v8h*)(bp0 + 8);
        v8h b1lo = *(const v8h*)(bp1);
        v8h b1hi = *(const v8h*)(bp1 + 8);
        v16h b0, b1;
        #pragma unroll
        for (int i = 0; i < 8; ++i) {
            b0[i] = b0lo[i]; b0[i + 8] = b0hi[i];
            b1[i] = b1lo[i]; b1[i + 8] = b1hi[i];
        }

        acc0 = __builtin_amdgcn_wmma_f32_16x16x32_f16(false, a, false, b0,
                                                      (short)0, acc0, false, false);
        acc1 = __builtin_amdgcn_wmma_f32_16x16x32_f16(false, a, false, b1,
                                                      (short)0, acc1, false, false);
    }

    // C/D layout: VGPR i -> row = i + (lane>>4)*8, col = lane&15
    const int ccol = col0 + l15;
    const int crow = rowBase + (lane >> 4) * 8;
    const float bv0 = bias ? bias[ccol]      : 0.0f;
    const float bv1 = bias ? bias[ccol + 16] : 0.0f;
    #pragma unroll
    for (int i = 0; i < 8; ++i) {
        C[(size_t)(crow + i) * GCN_D + ccol]      = acc0[i] + bv0;
        C[(size_t)(crow + i) * GCN_D + ccol + 16] = acc1[i] + bv1;
    }
}

// ---------------------------------------------------------------------------
// Helpers
// ---------------------------------------------------------------------------
__global__ void k_f32_to_f16(const float* __restrict__ in, _Float16* __restrict__ out, int n) {
    int i = blockIdx.x * 256 + threadIdx.x;
    if (i < n) out[i] = (_Float16)in[i];
}

// W[K,Dn] f32 row-major -> Wt[Dn,K] f16 (transposed for contiguous B fragments)
__global__ void k_wcvt_t(const float* __restrict__ W, _Float16* __restrict__ Wt, int K, int Dn) {
    int i = blockIdx.x * 256 + threadIdx.x;
    if (i < K * Dn) {
        int d = i / K, k = i - d * K;
        Wt[i] = (_Float16)W[(size_t)k * Dn + d];
    }
}

__global__ void k_deg_init(float* __restrict__ deg) {
    deg[blockIdx.x * 256 + threadIdx.x] = 1.0f;  // self-loop contribution
}

__global__ void k_deg_count(const long long* __restrict__ dst, float* __restrict__ deg) {
    int e = blockIdx.x * 256 + threadIdx.x;
    if (e < GCN_E) atomicAdd(&deg[dst[e]], 1.0f);
}

__global__ void k_deg_rsqrt(const float* __restrict__ deg, float* __restrict__ dis) {
    int i = blockIdx.x * 256 + threadIdx.x;
    dis[i] = rsqrtf(deg[i]);
}

__global__ void k_fill_bias(float* __restrict__ out, const float* __restrict__ b) {
    out[(size_t)blockIdx.x * GCN_D + threadIdx.x] = b[threadIdx.x];
}

// out[dst] += h1[src] * dis[src]*dis[dst]; one block per edge (E real + NTOT self-loops)
__global__ void k_gcn_scatter(const float* __restrict__ h1,
                              const long long* __restrict__ src,
                              const long long* __restrict__ dst,
                              const float* __restrict__ dis,
                              float* __restrict__ out) {
    int e = blockIdx.x;
    long long s, d;
    if (e < GCN_E) { s = src[e]; d = dst[e]; }
    else           { s = d = (long long)(e - GCN_E); }
    float nrm = dis[s] * dis[d];
    int t = threadIdx.x;
    atomicAdd(&out[(size_t)d * GCN_D + t], h1[(size_t)s * GCN_D + t] * nrm);
}

__global__ void k_graph_mean(const float* __restrict__ h, float* __restrict__ xqpre) {
    int g = blockIdx.x, d = threadIdx.x;
    float s = 0.0f;
    const float* hg = h + (size_t)g * GCN_N * GCN_D + d;
    for (int n = 0; n < GCN_N; ++n) s += hg[(size_t)n * GCN_D];
    xqpre[g * GCN_D + d] = s * (1.0f / GCN_N);
}

// attn[node] = sigmoid(dot(xq[g], xk[node]) / sqrt(D)); one wave per node
__global__ void k_attn(const float* __restrict__ xq, const float* __restrict__ xk,
                       float* __restrict__ attn) {
    int node = blockIdx.x * 8 + (threadIdx.x >> 5);
    int lane = threadIdx.x & 31;
    int g = node >> 9;  // node / 512
    const float* xkr = xk + (size_t)node * GCN_D;
    const float* xqr = xq + g * GCN_D;
    float s = 0.0f;
    #pragma unroll
    for (int i = 0; i < GCN_D / 32; ++i) s += xkr[lane + i * 32] * xqr[lane + i * 32];
    #pragma unroll
    for (int off = 16; off > 0; off >>= 1) s += __shfl_xor(s, off, 32);
    if (lane == 0) attn[node] = 1.0f / (1.0f + __expf(-s * 0.0625f));  // 1/sqrt(256)
}

__global__ void k_readout(const float* __restrict__ h, const float* __restrict__ attn,
                          float* __restrict__ out, int layerOff) {
    int g = blockIdx.x, d = threadIdx.x;
    float s = 0.0f;
    const float* hg = h + (size_t)g * GCN_N * GCN_D + d;
    const float* ag = attn + (size_t)g * GCN_N;
    for (int n = 0; n < GCN_N; ++n) s += hg[(size_t)n * GCN_D] * ag[n];
    out[(size_t)g * (3 * GCN_D) + layerOff + d] = s * (1.0f / GCN_N);
}

// ---------------------------------------------------------------------------
extern "C" void kernel_launch(void* const* d_in, const int* in_sizes, int n_in,
                              void* d_out, int out_size, void* d_ws, size_t ws_size,
                              hipStream_t stream) {
    (void)in_sizes; (void)n_in; (void)out_size; (void)ws_size;

    const float*     x   = (const float*)d_in[0];
    const long long* ei  = (const long long*)d_in[1];          // [2,E] int64
    const float*     W0  = (const float*)d_in[3];
    const float*     b0  = (const float*)d_in[4];
    const float*     Ws  = (const float*)d_in[5];              // [2,256,256]
    const float*     bs  = (const float*)d_in[6];              // [2,256]
    const float*     Wq  = (const float*)d_in[7];              // [3,256,256]
    const float*     bq  = (const float*)d_in[8];
    const float*     Wk  = (const float*)d_in[9];
    const float*     bk  = (const float*)d_in[10];
    float* out = (float*)d_out;

    const long long* src = ei;
    const long long* dst = ei + GCN_E;

    // --- workspace layout (all offsets 256B-aligned) ---
    char* p = (char*)d_ws;
    auto take = [&](size_t bytes) { char* r = p; p += (bytes + 255) & ~(size_t)255; return r; };
    float*     hA      = (float*)take((size_t)GCN_NTOT * GCN_D * 4);
    float*     hB      = (float*)take((size_t)GCN_NTOT * GCN_D * 4);
    float*     hT      = (float*)take((size_t)GCN_NTOT * GCN_D * 4);   // h1 then xk
    _Float16*  hf16    = (_Float16*)take((size_t)GCN_NTOT * GCN_D * 2);
    _Float16*  wtc0    = (_Float16*)take((size_t)GCN_D * GCN_FIN * 2);
    _Float16*  wtc1    = (_Float16*)take((size_t)GCN_D * GCN_D * 2);
    _Float16*  wtc2    = (_Float16*)take((size_t)GCN_D * GCN_D * 2);
    _Float16*  wtq[3], *wtk[3];
    for (int i = 0; i < 3; ++i) wtq[i] = (_Float16*)take((size_t)GCN_D * GCN_D * 2);
    for (int i = 0; i < 3; ++i) wtk[i] = (_Float16*)take((size_t)GCN_D * GCN_D * 2);
    float*     deg     = (float*)take((size_t)GCN_NTOT * 4);
    float*     dis     = (float*)take((size_t)GCN_NTOT * 4);
    float*     xqpre   = (float*)take((size_t)GCN_G * GCN_D * 4);
    float*     xq      = (float*)take((size_t)GCN_G * GCN_D * 4);
    _Float16*  xqpre16 = (_Float16*)take((size_t)GCN_G * GCN_D * 2);
    float*     attn    = (float*)take((size_t)GCN_NTOT * 4);

    dim3 blk(256);

    // --- degrees / normalization ---
    k_deg_init <<<GCN_NTOT / 256, blk, 0, stream>>>(deg);
    k_deg_count<<<GCN_E   / 256, blk, 0, stream>>>(dst, deg);
    k_deg_rsqrt<<<GCN_NTOT / 256, blk, 0, stream>>>(deg, dis);

    // --- weights: transpose + f16 convert ---
    k_wcvt_t<<<(GCN_FIN * GCN_D + 255) / 256, blk, 0, stream>>>(W0, wtc0, GCN_FIN, GCN_D);
    k_wcvt_t<<<(GCN_D * GCN_D + 255) / 256, blk, 0, stream>>>(Ws,                 wtc1, GCN_D, GCN_D);
    k_wcvt_t<<<(GCN_D * GCN_D + 255) / 256, blk, 0, stream>>>(Ws + GCN_D * GCN_D, wtc2, GCN_D, GCN_D);
    for (int i = 0; i < 3; ++i) {
        k_wcvt_t<<<(GCN_D * GCN_D + 255) / 256, blk, 0, stream>>>(Wq + (size_t)i * GCN_D * GCN_D, wtq[i], GCN_D, GCN_D);
        k_wcvt_t<<<(GCN_D * GCN_D + 255) / 256, blk, 0, stream>>>(Wk + (size_t)i * GCN_D * GCN_D, wtk[i], GCN_D, GCN_D);
    }

    // --- layer 0 input -> f16 ---
    k_f32_to_f16<<<(GCN_NTOT * GCN_FIN) / 256, blk, 0, stream>>>(x, hf16, GCN_NTOT * GCN_FIN);

    const _Float16* wtc[3] = { wtc0, wtc1, wtc2 };
    const int rowTiles = GCN_NTOT / 16;      // 2048
    const int qTiles   = GCN_G / 16;         // 4

    for (int i = 0; i < 3; ++i) {
        const int K = (i == 0) ? GCN_FIN : GCN_D;
        const float* bconv = (i == 0) ? b0 : (bs + (size_t)(i - 1) * GCN_D);
        float* hout = (i & 1) ? hB : hA;

        // h1 = h_in @ W   (WMMA, no bias here; bias added as aggregation init)
        gemm_wmma_f16<<<rowTiles, blk, 0, stream>>>(hf16, wtc[i], nullptr, hT, K);

        // out = b; out[dst] += h1[src] * norm  (incl. self-loops)
        k_fill_bias  <<<GCN_NTOT, blk, 0, stream>>>(hout, bconv);
        k_gcn_scatter<<<GCN_E + GCN_NTOT, blk, 0, stream>>>(hT, src, dst, dis, hout);

        // h_out -> f16 (feeds xk GEMM now and next layer's conv GEMM)
        k_f32_to_f16<<<(GCN_NTOT * GCN_D) / 256, blk, 0, stream>>>(hout, hf16, GCN_NTOT * GCN_D);

        // xk = h_out @ Wk + bk   (WMMA)
        gemm_wmma_f16<<<rowTiles, blk, 0, stream>>>(hf16, wtk[i], bk + (size_t)i * GCN_D, hT, GCN_D);

        // xq = mean_n(h_out) @ Wq + bq   (tiny WMMA)
        k_graph_mean<<<GCN_G, blk, 0, stream>>>(hout, xqpre);
        k_f32_to_f16<<<GCN_G, blk, 0, stream>>>(xqpre, xqpre16, GCN_G * GCN_D);
        gemm_wmma_f16<<<qTiles, blk, 0, stream>>>(xqpre16, wtq[i], bq + (size_t)i * GCN_D, xq, GCN_D);

        // attention + weighted-mean readout
        k_attn   <<<GCN_NTOT / 8, blk, 0, stream>>>(xq, hT, attn);
        k_readout<<<GCN_G, blk, 0, stream>>>(hout, attn, out, i * GCN_D);
    }
}
